// OptionB_48455821033917
// MI455X (gfx1250) — compile-verified, matrix operand order checked by reading, not run on hardware
//
#include <hip/hip_runtime.h>
#include <math.h>

// ---------------- static config (must match reference) ----------------
#define B_     64
#define NL_    96
#define NR_    256
#define H_     512
#define HEADS_ 8
#define DH_    64
#define CAH_   8
#define EL_G_  36864     /* 576*64 ligand edges total  */
#define ER_G_  196608    /* 3072*64 residue edges total */
#define NLT_   6144      /* B*NL */
#define NRT_   16384     /* B*NR */
#define CONDD_ 33
#define EPS_   1e-5f

typedef _Float16 half_t;
typedef __attribute__((ext_vector_type(2)))  _Float16 v2h;
typedef __attribute__((ext_vector_type(8)))  _Float16 v8h;
typedef __attribute__((ext_vector_type(16))) _Float16 v16h;
typedef __attribute__((ext_vector_type(2)))  float    v2f;
typedef __attribute__((ext_vector_type(8)))  float    v8f;

// ---------------- small device helpers ----------------
__device__ __forceinline__ float gelu_f(float x) {
    return 0.5f * x * (1.0f + erff(x * 0.70710678118654752f));
}
__device__ __forceinline__ void atomicMaxF(float* addr, float val) {
    unsigned int* ai = (unsigned int*)addr;
    unsigned int old = *ai, assumed;
    while (__uint_as_float(old) < val) {
        assumed = old;
        old = atomicCAS(ai, assumed, __float_as_uint(val));
        if (old == assumed) break;
    }
}

// ---------------- generic fill ----------------
__global__ void fill_kernel(float* __restrict__ p, float v, long n) {
    long t = (long)blockIdx.x * blockDim.x + threadIdx.x;
    if (t < n) p[t] = v;
}

// ---------------- WMMA GEMM: C[M,N] = act(A[M,K] @ B[K,N] + bias) + Radd ----
// fp32 in/out, f16 WMMA 16x16x32 with f32 accumulation.
// Block: 128 threads (4 waves). Block tile 128(M) x 64(N), K-step 32.
// Wave w owns 32 rows -> 2 A-fragments x 4 B-fragments = 8 WMMAs / K-step.
// LDS: A row-major, B *transposed* ([col][k]); row stride 40 halves (80 B) so
// every fragment is two aligned 16-byte runs -> ds_load_b128 pairs.
// Staging: uniform fast path (branch-free vectorized loads) for interior
// tiles; guarded slow path only for M/N/K edge tiles.
#define GTM 128
#define GTN 64
#define GTK 32
#define LSTR 40   /* LDS row stride in halves; 80 B, multiple of 16 B */
__global__ __launch_bounds__(128)
void gemm_wmma(const float* __restrict__ A, const float* __restrict__ Bw,
               const float* __restrict__ bias, const float* __restrict__ Radd,
               float* __restrict__ C, int M, int N, int K, int act)
{
    __shared__ __align__(16) half_t As[GTM][LSTR];  // 10240 B
    __shared__ __align__(16) half_t Bs[GTN][LSTR];  // 5120 B (transposed tile)
    const int tid  = threadIdx.x;
    const int wave = tid >> 5;
    const int lane = tid & 31;
    const int bm = blockIdx.y * GTM;
    const int bn = blockIdx.x * GTN;

    v8f acc[2][4] = {};

    const int hs = lane >> 4;     // 0: lanes 0-15, 1: lanes 16-31
    const int lr = lane & 15;
    const int kb = hs * 8;        // K sub-offset per ISA 16-bit layout

    const bool mFull = (bm + GTM <= M);
    const bool nFull = (bn + GTN <= N);
    const bool kEven = ((K & 1) == 0);

    for (int k0 = 0; k0 < K; k0 += GTK) {
        const bool kFull = (k0 + GTK <= K);
        // ---- stage A tile (128 x 32) as packed half2 pairs along K ----
        if (mFull && kFull && kEven) {
            #pragma unroll
            for (int i = tid; i < GTM * (GTK / 2); i += 128) {
                int r  = i >> 4;          // row 0..127
                int c2 = i & 15;          // K-pair index
                v2f f = *(const v2f*)&A[(size_t)(bm + r) * K + k0 + 2 * c2];
                v2h p; p[0] = (half_t)f[0]; p[1] = (half_t)f[1];
                *(v2h*)&As[r][2 * c2] = p;
            }
        } else {
            for (int i = tid; i < GTM * (GTK / 2); i += 128) {
                int r  = i >> 4;
                int c2 = i & 15;
                int gr = bm + r, gc = k0 + 2 * c2;
                float f0 = (gr < M && gc     < K) ? A[(size_t)gr * K + gc]     : 0.0f;
                float f1 = (gr < M && gc + 1 < K) ? A[(size_t)gr * K + gc + 1] : 0.0f;
                v2h p; p[0] = (half_t)f0; p[1] = (half_t)f1;
                *(v2h*)&As[r][2 * c2] = p;
            }
        }
        // ---- stage B tile (32 x 64) transposed: Bs[col][k] ----
        if (nFull && kFull) {
            #pragma unroll
            for (int i = tid; i < (GTK / 2) * GTN; i += 128) {
                int c = i & 63;           // column (fastest -> coalesced)
                int p = i >> 6;           // K-pair index
                const float* bp = &Bw[(size_t)(k0 + 2 * p) * N + bn + c];
                float f0 = bp[0];
                float f1 = bp[N];
                v2h pk; pk[0] = (half_t)f0; pk[1] = (half_t)f1;
                *(v2h*)&Bs[c][2 * p] = pk;
            }
        } else {
            for (int i = tid; i < (GTK / 2) * GTN; i += 128) {
                int c = i & 63;
                int p = i >> 6;
                int gk = k0 + 2 * p, gc = bn + c;
                float f0 = (gk     < K && gc < N) ? Bw[(size_t)gk * N + gc]       : 0.0f;
                float f1 = (gk + 1 < K && gc < N) ? Bw[(size_t)(gk + 1) * N + gc] : 0.0f;
                v2h pk; pk[0] = (half_t)f0; pk[1] = (half_t)f1;
                *(v2h*)&Bs[c][2 * p] = pk;
            }
        }
        __syncthreads();

        // A fragments: halves [0..7]=K kb..kb+7, [8..15]=K kb+16..kb+23 (ISA table order)
        v16h af[2];
        #pragma unroll
        for (int m = 0; m < 2; ++m) {
            int arow = wave * 32 + m * 16 + lr;
            v8h lo = *(const v8h*)&As[arow][kb];
            v8h hi = *(const v8h*)&As[arow][kb + 16];
            af[m] = __builtin_shufflevector(lo, hi, 0,1,2,3,4,5,6,7,8,9,10,11,12,13,14,15);
        }
        #pragma unroll
        for (int nt = 0; nt < 4; ++nt) {
            int bcol = nt * 16 + lr;
            v8h lo = *(const v8h*)&Bs[bcol][kb];
            v8h hi = *(const v8h*)&Bs[bcol][kb + 16];
            v16h bf = __builtin_shufflevector(lo, hi, 0,1,2,3,4,5,6,7,8,9,10,11,12,13,14,15);
            acc[0][nt] = __builtin_amdgcn_wmma_f32_16x16x32_f16(
                false, af[0], false, bf, (short)0, acc[0][nt], false, false);
            acc[1][nt] = __builtin_amdgcn_wmma_f32_16x16x32_f16(
                false, af[1], false, bf, (short)0, acc[1][nt], false, false);
        }
        __syncthreads();
    }

    // store: C/D layout — lane (l&15)=col, VGPR r -> row r + 8*(l>=16)
    #pragma unroll
    for (int m = 0; m < 2; ++m) {
        #pragma unroll
        for (int nt = 0; nt < 4; ++nt) {
            int gc = bn + nt * 16 + lr;
            if (gc >= N) continue;
            #pragma unroll
            for (int r = 0; r < 8; ++r) {
                int gr = bm + wave * 32 + m * 16 + hs * 8 + r;
                if (gr >= M) continue;
                float v = acc[m][nt][r];
                if (bias) v += bias[gc];
                if (act == 1)      v = gelu_f(v);
                else if (act == 2) v = fmaxf(v, 0.0f);
                if (Radd) v += Radd[(size_t)gr * N + gc];
                C[(size_t)gr * N + gc] = v;
            }
        }
    }
}

// ---------------- GAT segment kernels ----------------
// edge-attr mean by dst (self-loop fill)
__global__ void ea_accum(const float* __restrict__ ea, const int* __restrict__ dst,
                         float* __restrict__ eam, float* __restrict__ deg, int E, int ed)
{
    int e = blockIdx.x * blockDim.x + threadIdx.x;
    if (e >= E) return;
    int d = dst[e];
    atomicAdd(&deg[d], 1.0f);
    for (int c = 0; c < ed; ++c)
        atomicAdd(&eam[(size_t)d * ed + c], ea[(size_t)e * ed + c]);
}
__global__ void ea_div(float* __restrict__ eam, const float* __restrict__ deg,
                       int n, int ed)
{
    long t = (long)blockIdx.x * blockDim.x + threadIdx.x;
    if (t >= (long)n * ed) return;
    eam[t] /= fmaxf(deg[t / ed], 1.0f);
}

// per (edge, head) attention logit + scatter-max.  Edges [0,E): real, [E,E+n): self-loops.
template <int ED>
__global__ void gat_logits(const float* __restrict__ xl, const float* __restrict__ xr,
                           const float* __restrict__ ea, const float* __restrict__ eam,
                           const int* __restrict__ ei, const float* __restrict__ We,
                           const float* __restrict__ att, float* __restrict__ logit,
                           float* __restrict__ mx, int E, int n)
{
    long t = (long)blockIdx.x * blockDim.x + threadIdx.x;
    if (t >= (long)(E + n) * HEADS_) return;
    int e = (int)(t >> 3), h = (int)(t & 7);
    int s, d_;
    const float* eap;
    if (e < E) { s = ei[e]; d_ = ei[E + e]; eap = ea + (size_t)e * ED; }
    else       { s = e - E; d_ = e - E;     eap = eam + (size_t)(e - E) * ED; }
    float eav[ED];
    #pragma unroll
    for (int c = 0; c < ED; ++c) eav[c] = eap[c];
    const float* xlp = xl + (size_t)s  * H_ + h * DH_;
    const float* xrp = xr + (size_t)d_ * H_ + h * DH_;
    const float* ap  = att + h * DH_;
    float lg = 0.0f;
    for (int d0 = 0; d0 < DH_; ++d0) {
        float ew = 0.0f;
        #pragma unroll
        for (int c = 0; c < ED; ++c) ew += eav[c] * We[(size_t)c * H_ + h * DH_ + d0];
        float m = xlp[d0] + xrp[d0] + ew;
        m = (m > 0.0f) ? m : 0.2f * m;          // leaky_relu(0.2)
        lg += m * ap[d0];
    }
    logit[t] = lg;
    atomicMaxF(&mx[(size_t)d_ * HEADS_ + h], lg);
}

__global__ void gat_expsum(float* __restrict__ logit, const float* __restrict__ mx,
                           float* __restrict__ den, const int* __restrict__ ei,
                           int E, int n)
{
    long t = (long)blockIdx.x * blockDim.x + threadIdx.x;
    if (t >= (long)(E + n) * HEADS_) return;
    int e = (int)(t >> 3), h = (int)(t & 7);
    int d_ = (e < E) ? ei[E + e] : (e - E);
    float ex = expf(logit[t] - mx[(size_t)d_ * HEADS_ + h]);
    logit[t] = ex;
    atomicAdd(&den[(size_t)d_ * HEADS_ + h], ex);
}

__global__ void gat_aggregate(const float* __restrict__ ex, const float* __restrict__ den,
                              const float* __restrict__ xl, const int* __restrict__ ei,
                              float* __restrict__ out, int E, int n)
{
    long t = (long)blockIdx.x * blockDim.x + threadIdx.x;
    if (t >= (long)(E + n) * HEADS_) return;
    int e = (int)(t >> 3), h = (int)(t & 7);
    int s, d_;
    if (e < E) { s = ei[e]; d_ = ei[E + e]; } else { s = e - E; d_ = e - E; }
    float a = ex[t] / den[(size_t)d_ * HEADS_ + h];
    const float* xs = xl + (size_t)s * H_ + h * DH_;
    float* op = out + (size_t)d_ * H_ + h * DH_;
    for (int d0 = 0; d0 < DH_; ++d0) atomicAdd(&op[d0], a * xs[d0]);
}

// ---------------- misc elementwise / norms ----------------
__global__ void add_bias(float* __restrict__ x, const float* __restrict__ b,
                         long total, int C)
{
    long t = (long)blockIdx.x * blockDim.x + threadIdx.x;
    if (t < total) x[t] += b[t % C];
}
__global__ void bn_accum(const float* __restrict__ x, float* __restrict__ st,
                         long total, int C)
{
    long t = (long)blockIdx.x * blockDim.x + threadIdx.x;
    if (t >= total) return;
    float v = x[t];
    int c = (int)(t % C);
    atomicAdd(&st[c], v);
    atomicAdd(&st[C + c], v * v);
}
__global__ void bn_apply(const float* __restrict__ x, const float* __restrict__ st,
                         const float* __restrict__ g, const float* __restrict__ bb,
                         const float* __restrict__ skip, float* __restrict__ y,
                         int n, int C)
{
    long t = (long)blockIdx.x * blockDim.x + threadIdx.x;
    if (t >= (long)n * C) return;
    int c = (int)(t % C);
    float mu  = st[c] / (float)n;
    float var = st[C + c] / (float)n - mu * mu;
    float v = (x[t] - mu) * rsqrtf(var + EPS_) * g[c] + bb[c];
    y[t] = gelu_f(v) + skip[t];
}
__global__ void head_mean(const float* __restrict__ in, const float* __restrict__ bias,
                          float* __restrict__ out, int n)
{
    long t = (long)blockIdx.x * blockDim.x + threadIdx.x;
    if (t >= (long)n * DH_) return;
    int i = (int)(t / DH_), d = (int)(t % DH_);
    float s = 0.0f;
    #pragma unroll
    for (int h = 0; h < HEADS_; ++h) s += in[(size_t)i * H_ + h * DH_ + d];
    out[t] = s * (1.0f / HEADS_) + bias[d];
}

// ---------------- cross attention: one wave per (b,h,l) ----------------
__global__ __launch_bounds__(32)
void cross_attn(const float* __restrict__ Q, const float* __restrict__ Kb,
                const float* __restrict__ Vb, float* __restrict__ ctx)
{
    int blk = blockIdx.x;
    int l = blk % NL_; blk /= NL_;
    int h = blk % CAH_; blk /= CAH_;
    int b = blk;
    int lane = threadIdx.x;
    const float scale = 0.125f;   // 64^-0.5
    const float* qp = Q + ((size_t)(b * NL_ + l)) * H_ + h * DH_;
    float logit[8];
    #pragma unroll
    for (int j = 0; j < 8; ++j) {
        int r = lane + 32 * j;
        const float* kp = Kb + ((size_t)(b * NR_ + r)) * H_ + h * DH_;
        float s = 0.0f;
        for (int d = 0; d < DH_; ++d) s += qp[d] * kp[d];
        logit[j] = s * scale;
    }
    float mx = logit[0];
    #pragma unroll
    for (int j = 1; j < 8; ++j) mx = fmaxf(mx, logit[j]);
    for (int o = 16; o > 0; o >>= 1) mx = fmaxf(mx, __shfl_xor(mx, o, 32));
    float w[8], se = 0.0f;
    #pragma unroll
    for (int j = 0; j < 8; ++j) { w[j] = expf(logit[j] - mx); se += w[j]; }
    for (int o = 16; o > 0; o >>= 1) se += __shfl_xor(se, o, 32);
    float inv = 1.0f / se;
    float acc[DH_];
    for (int d = 0; d < DH_; ++d) acc[d] = 0.0f;
    #pragma unroll
    for (int j = 0; j < 8; ++j) {
        int r = lane + 32 * j;
        const float* vp = Vb + ((size_t)(b * NR_ + r)) * H_ + h * DH_;
        float a = w[j] * inv;
        for (int d = 0; d < DH_; ++d) acc[d] += a * vp[d];
    }
    float* cp = ctx + ((size_t)(b * NL_ + l)) * H_ + h * DH_;
    for (int d = 0; d < DH_; ++d) {
        float v = acc[d];
        for (int o = 16; o > 0; o >>= 1) v += __shfl_xor(v, o, 32);
        if (lane == 0) cp[d] = v;
    }
}

__global__ __launch_bounds__(32)
void layernorm_rows(const float* __restrict__ x, const float* __restrict__ g,
                    const float* __restrict__ bta, float* __restrict__ y, int C)
{
    int row = blockIdx.x, lane = threadIdx.x;
    const float* xp = x + (size_t)row * C;
    float s = 0.0f, s2 = 0.0f;
    for (int c = lane; c < C; c += 32) { float v = xp[c]; s += v; s2 += v * v; }
    for (int o = 16; o > 0; o >>= 1) { s += __shfl_xor(s, o, 32); s2 += __shfl_xor(s2, o, 32); }
    float mu = s / C, var = s2 / C - mu * mu;
    float inv = rsqrtf(var + EPS_);
    float* yp = y + (size_t)row * C;
    for (int c = lane; c < C; c += 32) yp[c] = (xp[c] - mu) * inv * g[c] + bta[c];
}

__global__ void mean_pool(const float* __restrict__ q, float* __restrict__ hlig)
{
    long t = (long)blockIdx.x * blockDim.x + threadIdx.x;
    if (t >= (long)B_ * H_) return;
    int b = (int)(t / H_), c = (int)(t % H_);
    float s = 0.0f;
    for (int l = 0; l < NL_; ++l) s += q[((size_t)(b * NL_ + l)) * H_ + c];
    hlig[t] = s * (1.0f / NL_);
}

__global__ void build_cond(const float* __restrict__ ph, const float* __restrict__ tp,
                           const float* __restrict__ flag, const float* __restrict__ be,
                           const int* __restrict__ bidx, float* __restrict__ cond)
{
    int b = blockIdx.x * blockDim.x + threadIdx.x;
    if (b >= B_) return;
    float* cp = cond + (size_t)b * CONDD_;
    for (int i = 0; i < 8; ++i)  cp[i]      = ph[b * 8 + i];
    for (int i = 0; i < 8; ++i)  cp[8 + i]  = tp[b * 8 + i];
    int bi = bidx[b];
    for (int i = 0; i < 16; ++i) cp[16 + i] = be[bi * 16 + i];
    cp[32] = flag[b];
}

__global__ void concat_hc(const float* __restrict__ hlig, const float* __restrict__ hc,
                          float* __restrict__ hcat)
{
    long t = (long)blockIdx.x * blockDim.x + threadIdx.x;
    if (t >= (long)B_ * (H_ + 64)) return;
    int b = (int)(t / (H_ + 64)), c = (int)(t % (H_ + 64));
    hcat[t] = (c < H_) ? hlig[(size_t)b * H_ + c] : hc[(size_t)b * 64 + (c - H_)];
}

// ---------------- host-side helpers ----------------
static inline void fillk(hipStream_t s, float* p, float v, long n) {
    fill_kernel<<<(unsigned)((n + 255) / 256), 256, 0, s>>>(p, v, n);
}
static inline void run_gemm(hipStream_t s, const float* A, const float* Bw,
                            const float* bias, const float* radd, float* C,
                            int M, int N, int K, int act) {
    dim3 g((N + GTN - 1) / GTN, (M + GTM - 1) / GTM);
    gemm_wmma<<<g, 128, 0, s>>>(A, Bw, bias, radd, C, M, N, K, act);
}

// One GNN stack.  Param leaf layout (pytree-sorted within each layer dict):
//   layer: We(+0) Wl(+1) Wr(+2) att(+3) bias(+4) bl(+5) bn_b(+6) bn_g(+7) br(+8) [skip_W(+9)]
//   stack: layers0 @base, layers1 @base+9, layers2 @base+18; post_W base+28, post_b +29,
//          proj_W +30, proj_b +31.
static float* run_stack(hipStream_t s, const float* const* P, int base,
                        const float* xin, int n, int ind,
                        const int* ei, const float* ea, int E, int ed,
                        float* bx, float* btmp, float* bxl, float* bxr,
                        float* eam, float* deg, float* logit, float* mx,
                        float* den, float* bns)
{
    run_gemm(s, xin, P[base + 30], P[base + 31], nullptr, bx, n, H_, ind, 0);

    fillk(s, deg, 0.0f, n);
    fillk(s, eam, 0.0f, (long)n * ed);
    ea_accum<<<(E + 255) / 256, 256, 0, s>>>(ea, ei + E, eam, deg, E, ed);
    ea_div<<<(unsigned)(((long)n * ed + 255) / 256), 256, 0, s>>>(eam, deg, n, ed);

    float* cx = bx; float* ct = btmp;
    const long tt = (long)(E + n) * HEADS_;
    const unsigned tb = (unsigned)((tt + 255) / 256);
    for (int i = 0; i < 3; ++i) {
        int li = base + ((i == 2) ? 18 : 9 * i);
        const float *We = P[li + 0], *Wl = P[li + 1], *Wr = P[li + 2], *att = P[li + 3];
        const float *bias = P[li + 4], *bl = P[li + 5], *bn_b = P[li + 6];
        const float *bn_g = P[li + 7], *br = P[li + 8];

        run_gemm(s, cx, Wl, bl, nullptr, bxl, n, H_, H_, 0);
        run_gemm(s, cx, Wr, br, nullptr, bxr, n, H_, H_, 0);
        fillk(s, mx, -3.0e38f, (long)n * HEADS_);
        fillk(s, den, 0.0f, (long)n * HEADS_);
        if (ed == 16)
            gat_logits<16><<<tb, 256, 0, s>>>(bxl, bxr, ea, eam, ei, We, att, logit, mx, E, n);
        else
            gat_logits<32><<<tb, 256, 0, s>>>(bxl, bxr, ea, eam, ei, We, att, logit, mx, E, n);
        gat_expsum<<<tb, 256, 0, s>>>(logit, mx, den, ei, E, n);
        fillk(s, ct, 0.0f, (long)n * H_);
        gat_aggregate<<<tb, 256, 0, s>>>(logit, den, bxl, ei, ct, E, n);

        if (i < 2) {
            add_bias<<<(unsigned)(((long)n * H_ + 255) / 256), 256, 0, s>>>(ct, bias, (long)n * H_, H_);
            fillk(s, bns, 0.0f, 2 * H_);
            bn_accum<<<(unsigned)(((long)n * H_ + 255) / 256), 256, 0, s>>>(ct, bns, (long)n * H_, H_);
            bn_apply<<<(unsigned)(((long)n * H_ + 255) / 256), 256, 0, s>>>(ct, bns, bn_g, bn_b, cx, ct, n, H_);
            float* t_ = cx; cx = ct; ct = t_;
        } else {
            head_mean<<<(unsigned)(((long)n * DH_ + 255) / 256), 256, 0, s>>>(ct, bias, bxr, n);
            run_gemm(s, cx, P[li + 9], nullptr, nullptr, bxl, n, DH_, H_, 0);   // skip_W
            fillk(s, bns, 0.0f, 2 * DH_);
            bn_accum<<<(unsigned)(((long)n * DH_ + 255) / 256), 256, 0, s>>>(bxr, bns, (long)n * DH_, DH_);
            bn_apply<<<(unsigned)(((long)n * DH_ + 255) / 256), 256, 0, s>>>(bxr, bns, bn_g, bn_b, bxl, bxr, n, DH_);
            run_gemm(s, bxr, P[base + 28], P[base + 29], nullptr, cx, n, H_, DH_, 0);
        }
    }
    return cx;   // == bx after two swaps
}

// ---------------- entry point ----------------
// d_in layout (top level in setup_inputs() insertion order; 'params' pytree
// flattened JAX-style with dict keys sorted recursively -> 89 leaves):
//  0 box_embed | 1 cond_W1 | 2 cond_W2 | 3 cond_b1 | 4 cond_b2
//  5..11  cross[0]: Wk Wo Wq Wv bo ln_b ln_g ; 12..18 cross[1]
//  19..50 gnn_l (layers + post/proj as in run_stack comment)
//  51..82 gnn_p
//  83 mlp_W1 | 84 mlp_W2 | 85 mlp_W3 | 86 mlp_b1 | 87 mlp_b2 | 88 mlp_b3
//  89 x_lig | 90 ea_lig | 91 x_res | 92 ea_res | 93 ph_enc | 94 temp_enc
//  95 receptor_flag | 96 ei_lig | 97 ei_res | 98 box_idx
extern "C" void kernel_launch(void* const* d_in, const int* in_sizes, int n_in,
                              void* d_out, int out_size, void* d_ws, size_t ws_size,
                              hipStream_t stream)
{
    (void)in_sizes; (void)n_in; (void)out_size; (void)ws_size;
    const float* P[89];
    for (int i = 0; i < 89; ++i) P[i] = (const float*)d_in[i];
    const float* x_lig  = (const float*)d_in[89];
    const float* ea_lig = (const float*)d_in[90];
    const float* x_res  = (const float*)d_in[91];
    const float* ea_res = (const float*)d_in[92];
    const float* ph     = (const float*)d_in[93];
    const float* tp     = (const float*)d_in[94];
    const float* flag   = (const float*)d_in[95];
    const int*   ei_lig = (const int*)d_in[96];
    const int*   ei_res = (const int*)d_in[97];
    const int*   bidx   = (const int*)d_in[98];

    float* w = (float*)d_ws;
    size_t o = 0;
    auto alloc = [&](size_t nfl) { float* p = w + o; o += nfl; return p; };
    float* res_x  = alloc((size_t)NRT_ * H_);
    float* res_t  = alloc((size_t)NRT_ * H_);
    float* res_xl = alloc((size_t)NRT_ * H_);
    float* res_xr = alloc((size_t)NRT_ * H_);
    float* lig_x  = alloc((size_t)NLT_ * H_);
    float* lig_t  = alloc((size_t)NLT_ * H_);
    float* lig_xl = alloc((size_t)NLT_ * H_);
    float* lig_xr = alloc((size_t)NLT_ * H_);
    float* eam    = alloc((size_t)NRT_ * 32);
    float* deg    = alloc(NRT_);
    float* logit  = alloc((size_t)(ER_G_ + NRT_) * HEADS_);
    float* mx     = alloc((size_t)NRT_ * HEADS_);
    float* den    = alloc((size_t)NRT_ * HEADS_);
    float* bns    = alloc(2 * H_);
    float* hlig   = alloc((size_t)B_ * H_);
    float* cond   = alloc((size_t)B_ * CONDD_);
    float* hc1    = alloc((size_t)B_ * 64);
    float* hc2    = alloc((size_t)B_ * 64);
    float* hcat   = alloc((size_t)B_ * (H_ + 64));
    float* m1     = alloc((size_t)B_ * H_);
    float* m2     = alloc((size_t)B_ * 128);

    // GNN stacks (residue base=51, ligand base=19)
    float* resF = run_stack(stream, P, 51, x_res, NRT_, 128, ei_res, ea_res, ER_G_, 32,
                            res_x, res_t, res_xl, res_xr, eam, deg, logit, mx, den, bns);
    float* q    = run_stack(stream, P, 19, x_lig, NLT_, 64, ei_lig, ea_lig, EL_G_, 16,
                            lig_x, lig_t, lig_xl, lig_xr, eam, deg, logit, mx, den, bns);

    // cross attention x2
    for (int i = 0; i < 2; ++i) {
        int cb = 5 + 7 * i;   // Wk Wo Wq Wv bo ln_b ln_g
        run_gemm(stream, resF, P[cb + 0], nullptr, nullptr, res_xl, NRT_, H_, H_, 0); // K
        run_gemm(stream, resF, P[cb + 3], nullptr, nullptr, res_xr, NRT_, H_, H_, 0); // V
        run_gemm(stream, q,    P[cb + 2], nullptr, nullptr, lig_xl, NLT_, H_, H_, 0); // Q
        cross_attn<<<B_ * CAH_ * NL_, 32, 0, stream>>>(lig_xl, res_xl, res_xr, lig_xr);
        run_gemm(stream, lig_xr, P[cb + 1], P[cb + 4], q, lig_t, NLT_, H_, H_, 0);    // +bo +res
        layernorm_rows<<<NLT_, 32, 0, stream>>>(lig_t, P[cb + 6], P[cb + 5], q, H_);
    }

    // pool + conditioning + MLP head
    mean_pool<<<(B_ * H_ + 255) / 256, 256, 0, stream>>>(q, hlig);
    build_cond<<<1, B_, 0, stream>>>(ph, tp, flag, P[0], bidx, cond);
    run_gemm(stream, cond, P[1], P[3], nullptr, hc1, B_, 64, CONDD_, 2);  // relu
    run_gemm(stream, hc1,  P[2], P[4], nullptr, hc2, B_, 64, 64, 0);
    concat_hc<<<(B_ * (H_ + 64) + 255) / 256, 256, 0, stream>>>(hlig, hc2, hcat);
    run_gemm(stream, hcat, P[83], P[86], nullptr, m1, B_, H_, H_ + 64, 1); // gelu
    run_gemm(stream, m1,   P[84], P[87], nullptr, m2, B_, 128, H_, 1);     // gelu
    run_gemm(stream, m2,   P[85], P[88], nullptr, (float*)d_out, B_, 1, 128, 0);
}
